// CategorySpecificLinear_59820304499003
// MI455X (gfx1250) — matrix operand order repaired and checked by reading, not compiled, not run on hardware
//
#include <hip/hip_runtime.h>

typedef __attribute__((ext_vector_type(16))) __bf16 v16bf;
typedef __attribute__((ext_vector_type(8)))  float  v8f;
typedef __attribute__((ext_vector_type(4)))  unsigned int u32x4;
typedef __attribute__((ext_vector_type(8)))  int          i32x8;
typedef __attribute__((ext_vector_type(4)))  int          i32x4;

#define BM 128
#define BN 128
#define BK 32          // fallback kernel K step
#define BK2 64         // TDM kernel K step (128B rows)
#define S_DIM 512
#define I_DIM 1024
#define H_DIM 4096
#define NCAT 16
#define NBATCH 32

__device__ __forceinline__ unsigned short f32_to_bf16_rne(float f) {
    unsigned int u = __float_as_uint(f);
    unsigned int r = 0x7FFFu + ((u >> 16) & 1u);
    return (unsigned short)((u + r) >> 16);
}
__device__ __forceinline__ float bf16_to_f32(unsigned short h) {
    return __uint_as_float(((unsigned int)h) << 16);
}
__device__ __forceinline__ unsigned lds_off_of(const void* p) {
    // Generic LDS pointer = {SHARED_BASE aperture, offset[31:0]} -> low 32 bits are the LDS byte offset.
    return (unsigned)(reinterpret_cast<uintptr_t>(p));
}

union FragAB { v16bf v; unsigned int u[8]; };

// ---------------------------------------------------------------------------
// TDM helper: 2D tile load, 128 rows x 64 bf16 (128B) per row, row stride 1024
// elements. Descriptor per CDNA5 ISA.
// This toolchain's builtin takes 6 args (groups 0..3 + extra group + cpol);
// unused higher-dim groups are zero-filled (2D tile).
// ---------------------------------------------------------------------------
__device__ __forceinline__ void tdm_load_tile(unsigned lds_byte_off, const void* gptr) {
    unsigned long long ga = (unsigned long long)(uintptr_t)gptr;
    u32x4 g0;
    g0[0] = 1u;                                       // count=1, user descriptor
    g0[1] = lds_byte_off;                             // lds_addr
    g0[2] = (unsigned)(ga & 0xFFFFFFFFull);           // global_addr[31:0]
    g0[3] = (unsigned)((ga >> 32) & 0x1FFFFFFull)     // global_addr[56:32]
            | (2u << 30);                             // type = 2 ("image")
    // group1 fields: data_size=1 (2B), tensor_dim0/1 = big (never OOB),
    // tile_dim0=64, tile_dim1=128, tensor_dim0_stride=1024, rest 0.
    const unsigned TD = 0x7FFFFFu;
    i32x8 g1;
    g1[0] = (int)(1u << 16);                          // data_size = 2 bytes
    g1[1] = (int)((TD & 0xFFFFu) << 16);              // tensor_dim0[15:0]
    g1[2] = (int)((TD >> 16) | ((TD & 0xFFFFu) << 16));   // td0[31:16] | td1[15:0]
    g1[3] = (int)((TD >> 16) | ((unsigned)BK2 << 16));    // td1[31:16] | tile_dim0
    g1[4] = (int)BM;                                  // tile_dim1=128, tile_dim2=0
    g1[5] = (int)I_DIM;                               // tensor_dim0_stride lo32
    g1[6] = 0;                                        // stride0 hi | stride1 lo
    g1[7] = 0;                                        // stride1 hi
    i32x4 z4 = {0, 0, 0, 0};
    i32x8 z8 = {0, 0, 0, 0, 0, 0, 0, 0};
    __builtin_amdgcn_tensor_load_to_lds(g0, g1, z4, z4, z8, 0);
}

// ---------------------------------------------------------------------------
// Pre-pass 1: X (f32) -> Xhi/Xlo bf16 planes, same layout.
// ---------------------------------------------------------------------------
__global__ __launch_bounds__(256)
void convert_x_bf16x2(const float* __restrict__ X,
                      unsigned short* __restrict__ Xhi,
                      unsigned short* __restrict__ Xlo) {
    size_t i = ((size_t)blockIdx.x * 256 + threadIdx.x) * 4;
    const float4 v = *reinterpret_cast<const float4*>(X + i);
    float f[4] = {v.x, v.y, v.z, v.w};
    unsigned short h[4], l[4];
    #pragma unroll
    for (int e = 0; e < 4; ++e) {
        h[e] = f32_to_bf16_rne(f[e]);
        l[e] = f32_to_bf16_rne(f[e] - bf16_to_f32(h[e]));
    }
    *reinterpret_cast<ushort4*>(&Xhi[i]) = make_ushort4(h[0], h[1], h[2], h[3]);
    *reinterpret_cast<ushort4*>(&Xlo[i]) = make_ushort4(l[0], l[1], l[2], l[3]);
}

// ---------------------------------------------------------------------------
// Pre-pass 2: W[cat][k][n] (f32) -> transposed WhiT/WloT[cat][n][k] bf16.
// 32x32 LDS-tiled transpose.
// ---------------------------------------------------------------------------
__global__ __launch_bounds__(256)
void convert_w_transpose_bf16x2(const float* __restrict__ W,
                                unsigned short* __restrict__ WhiT,
                                unsigned short* __restrict__ WloT) {
    __shared__ unsigned short tHi[32][33];
    __shared__ unsigned short tLo[32][33];
    const int n0 = blockIdx.x * 32;
    const int k0 = blockIdx.y * 32;
    const int cat = blockIdx.z;
    const float* Wc = W + (size_t)cat * I_DIM * H_DIM;
    const int tid = threadIdx.x;

    #pragma unroll
    for (int p = 0; p < 4; ++p) {
        int idx = tid + p * 256;
        int r = idx >> 5, c = idx & 31;                 // r = k, c = n
        float v = Wc[(size_t)(k0 + r) * H_DIM + (n0 + c)];
        unsigned short h = f32_to_bf16_rne(v);
        tHi[r][c] = h;
        tLo[r][c] = f32_to_bf16_rne(v - bf16_to_f32(h));
    }
    __syncthreads();
    int r = tid >> 3;                                   // n-index in tile
    int c = (tid & 7) << 2;                             // k-index base
    size_t base = ((size_t)cat * H_DIM + (n0 + r)) * I_DIM + (k0 + c);
    *reinterpret_cast<ushort4*>(&WhiT[base]) =
        make_ushort4(tHi[c][r], tHi[c + 1][r], tHi[c + 2][r], tHi[c + 3][r]);
    *reinterpret_cast<ushort4*>(&WloT[base]) =
        make_ushort4(tLo[c][r], tLo[c + 1][r], tLo[c + 2][r], tLo[c + 3][r]);
}

// ---------------------------------------------------------------------------
// Main GEMM: TDM double-buffered bf16x3 WMMA from pre-converted planes.
// Block = 256 threads = 8 wave32 (2x4), tile 128x128, K step 64.
// ---------------------------------------------------------------------------
__global__ __launch_bounds__(256)
void cat_linear_wmma_tdm(const unsigned short* __restrict__ Xhi,
                         const unsigned short* __restrict__ Xlo,
                         const unsigned short* __restrict__ WhiT,
                         const unsigned short* __restrict__ WloT,
                         const int*   __restrict__ cat_ids,
                         const float* __restrict__ Bias,
                         float*       __restrict__ Out) {
    // [buf][plane][tile]: plane 0=A_hi 1=A_lo 2=B_hi 3=B_lo; 2*4*16KB = 128KB
    __shared__ __align__(16) unsigned short smem[2][4][BM * BK2];

    const int tid  = threadIdx.x;
    const int lane = tid & 31;
    const int wid  = tid >> 5;
    const int wm   = wid & 1;
    const int wn   = wid >> 1;
    const int g    = lane >> 4;
    const int lm   = lane & 15;

    const int n0 = blockIdx.x * BN;
    const int m0 = blockIdx.y * BM;
    const int bz = blockIdx.z;

    const int cat = cat_ids[bz];
    const unsigned short* Ah = Xhi  + ((size_t)bz  * S_DIM + m0) * I_DIM;
    const unsigned short* Al = Xlo  + ((size_t)bz  * S_DIM + m0) * I_DIM;
    const unsigned short* Bh = WhiT + ((size_t)cat * H_DIM + n0) * I_DIM;
    const unsigned short* Bl = WloT + ((size_t)cat * H_DIM + n0) * I_DIM;
    const float* bc = Bias + (size_t)cat * H_DIM;
    float*       Ob = Out  + (size_t)bz * S_DIM * H_DIM;

    v8f acc[4][2];
    #pragma unroll
    for (int mt = 0; mt < 4; ++mt)
        #pragma unroll
        for (int nt = 0; nt < 2; ++nt)
            acc[mt][nt] = (v8f){0.f, 0.f, 0.f, 0.f, 0.f, 0.f, 0.f, 0.f};

    const int NSTEP = I_DIM / BK2;   // 16

    if (wid == 0) {
        tdm_load_tile(lds_off_of(&smem[0][0][0]), Ah);
        tdm_load_tile(lds_off_of(&smem[0][1][0]), Al);
        tdm_load_tile(lds_off_of(&smem[0][2][0]), Bh);
        tdm_load_tile(lds_off_of(&smem[0][3][0]), Bl);
    }

    for (int t = 0; t < NSTEP; ++t) {
        const int buf = t & 1;
        if (wid == 0) {
            const int tn = (t + 1 < NSTEP) ? (t + 1) : t;   // last iter: harmless reload
            const size_t ko = (size_t)tn * BK2;
            tdm_load_tile(lds_off_of(&smem[buf ^ 1][0][0]), Ah + ko);
            tdm_load_tile(lds_off_of(&smem[buf ^ 1][1][0]), Al + ko);
            tdm_load_tile(lds_off_of(&smem[buf ^ 1][2][0]), Bh + ko);
            tdm_load_tile(lds_off_of(&smem[buf ^ 1][3][0]), Bl + ko);
            __builtin_amdgcn_s_wait_tensorcnt(4);           // tile t's 4 ops retired (in-order)
        }
        __syncthreads();

        const unsigned short* sAh = smem[buf][0];
        const unsigned short* sAl = smem[buf][1];
        const unsigned short* sBh = smem[buf][2];
        const unsigned short* sBl = smem[buf][3];

        #pragma unroll
        for (int ks = 0; ks < BK2; ks += 32) {
            FragAB a_hi[4], a_lo[4], b_hi[2], b_lo[2];
            #pragma unroll
            for (int mt = 0; mt < 4; ++mt) {
                int row = wm * 64 + mt * 16 + lm;
                #pragma unroll
                for (int j = 0; j < 8; ++j) {
                    int kk = ks + ((j < 4) ? (2 * j) : (16 + 2 * (j - 4))) + 8 * g;
                    a_hi[mt].u[j] = *reinterpret_cast<const unsigned int*>(&sAh[row * BK2 + kk]);
                    a_lo[mt].u[j] = *reinterpret_cast<const unsigned int*>(&sAl[row * BK2 + kk]);
                }
            }
            #pragma unroll
            for (int nt = 0; nt < 2; ++nt) {
                int col = wn * 32 + nt * 16 + lm;
                #pragma unroll
                for (int j = 0; j < 8; ++j) {
                    int kk = ks + ((j < 4) ? (2 * j) : (16 + 2 * (j - 4))) + 8 * g;
                    b_hi[nt].u[j] = *reinterpret_cast<const unsigned int*>(&sBh[col * BK2 + kk]);
                    b_lo[nt].u[j] = *reinterpret_cast<const unsigned int*>(&sBl[col * BK2 + kk]);
                }
            }
            #pragma unroll
            for (int mt = 0; mt < 4; ++mt) {
                #pragma unroll
                for (int nt = 0; nt < 2; ++nt) {
                    acc[mt][nt] = __builtin_amdgcn_wmma_f32_16x16x32_bf16(
                        false, a_hi[mt].v, false, b_hi[nt].v, (short)0, acc[mt][nt], false, false);
                    acc[mt][nt] = __builtin_amdgcn_wmma_f32_16x16x32_bf16(
                        false, a_hi[mt].v, false, b_lo[nt].v, (short)0, acc[mt][nt], false, false);
                    acc[mt][nt] = __builtin_amdgcn_wmma_f32_16x16x32_bf16(
                        false, a_lo[mt].v, false, b_hi[nt].v, (short)0, acc[mt][nt], false, false);
                }
            }
        }
        __syncthreads();
    }

    #pragma unroll
    for (int nt = 0; nt < 2; ++nt) {
        int n = n0 + wn * 32 + nt * 16 + lm;
        float bv = bc[n];
        #pragma unroll
        for (int mt = 0; mt < 4; ++mt) {
            #pragma unroll
            for (int j = 0; j < 8; ++j) {
                int m = m0 + wm * 64 + mt * 16 + 8 * g + j;
                Ob[(size_t)m * H_DIM + n] = acc[mt][nt][j] + bv;
            }
        }
    }
}

// ---------------------------------------------------------------------------
// Fallback: fused convert-in-kernel bf16x3 (round-1 kernel) for small ws_size.
// ---------------------------------------------------------------------------
__global__ __launch_bounds__(256)
void cat_linear_wmma_bf16x3(const float* __restrict__ X,
                            const int*   __restrict__ cat_ids,
                            const float* __restrict__ W,
                            const float* __restrict__ Bias,
                            float*       __restrict__ Out) {
    __shared__ unsigned short sA_hi[BM * BK];
    __shared__ unsigned short sA_lo[BM * BK];
    __shared__ unsigned short sB_hi[BN * BK];   // transposed [n][k]
    __shared__ unsigned short sB_lo[BN * BK];

    const int tid  = threadIdx.x;
    const int lane = tid & 31;
    const int wid  = tid >> 5;
    const int wm   = wid & 1;
    const int wn   = wid >> 1;
    const int g    = lane >> 4;
    const int lm   = lane & 15;

    const int n0 = blockIdx.x * BN;
    const int m0 = blockIdx.y * BM;
    const int bz = blockIdx.z;

    const int cat = cat_ids[bz];
    const float* Xb = X    + (size_t)bz  * S_DIM * I_DIM;
    const float* Wc = W    + (size_t)cat * I_DIM * H_DIM;
    const float* bc = Bias + (size_t)cat * H_DIM;
    float*       Ob = Out  + (size_t)bz  * S_DIM * H_DIM;

    v8f acc[4][2];
    #pragma unroll
    for (int mt = 0; mt < 4; ++mt)
        #pragma unroll
        for (int nt = 0; nt < 2; ++nt)
            acc[mt][nt] = (v8f){0.f, 0.f, 0.f, 0.f, 0.f, 0.f, 0.f, 0.f};

    for (int k0 = 0; k0 < I_DIM; k0 += BK) {
        __syncthreads();
        #pragma unroll
        for (int p = 0; p < 4; ++p) {
            int q   = tid + p * 256;
            int row = q >> 3;
            int c4  = (q & 7) << 2;
            const float4 v = *reinterpret_cast<const float4*>(
                Xb + (size_t)(m0 + row) * I_DIM + (k0 + c4));
            float f[4] = {v.x, v.y, v.z, v.w};
            unsigned short h[4], l[4];
            #pragma unroll
            for (int e = 0; e < 4; ++e) {
                h[e] = f32_to_bf16_rne(f[e]);
                l[e] = f32_to_bf16_rne(f[e] - bf16_to_f32(h[e]));
            }
            uint2 ph = make_uint2((unsigned)h[0] | ((unsigned)h[1] << 16),
                                  (unsigned)h[2] | ((unsigned)h[3] << 16));
            uint2 pl = make_uint2((unsigned)l[0] | ((unsigned)l[1] << 16),
                                  (unsigned)l[2] | ((unsigned)l[3] << 16));
            *reinterpret_cast<uint2*>(&sA_hi[row * BK + c4]) = ph;
            *reinterpret_cast<uint2*>(&sA_lo[row * BK + c4]) = pl;
        }
        #pragma unroll
        for (int p = 0; p < 4; ++p) {
            int q  = tid + p * 256;
            int kr = q >> 5;
            int c4 = (q & 31) << 2;
            const float4 v = *reinterpret_cast<const float4*>(
                Wc + (size_t)(k0 + kr) * H_DIM + (n0 + c4));
            float f[4] = {v.x, v.y, v.z, v.w};
            #pragma unroll
            for (int e = 0; e < 4; ++e) {
                unsigned short h = f32_to_bf16_rne(f[e]);
                unsigned short l = f32_to_bf16_rne(f[e] - bf16_to_f32(h));
                sB_hi[(c4 + e) * BK + kr] = h;
                sB_lo[(c4 + e) * BK + kr] = l;
            }
        }
        if (k0 + BK < I_DIM) {
            __builtin_prefetch(Xb + (size_t)(m0 + (tid >> 1)) * I_DIM + (k0 + BK) + ((tid & 1) << 4), 0, 1);
            __builtin_prefetch(Wc + (size_t)(k0 + BK + (tid >> 3)) * H_DIM + n0 + ((tid & 7) << 4), 0, 1);
        }
        __syncthreads();

        FragAB a_hi[4], a_lo[4], b_hi[2], b_lo[2];
        #pragma unroll
        for (int mt = 0; mt < 4; ++mt) {
            int row = wm * 64 + mt * 16 + lm;
            #pragma unroll
            for (int j = 0; j < 8; ++j) {
                int kk = ((j < 4) ? (2 * j) : (16 + 2 * (j - 4))) + 8 * g;
                a_hi[mt].u[j] = *reinterpret_cast<const unsigned int*>(&sA_hi[row * BK + kk]);
                a_lo[mt].u[j] = *reinterpret_cast<const unsigned int*>(&sA_lo[row * BK + kk]);
            }
        }
        #pragma unroll
        for (int nt = 0; nt < 2; ++nt) {
            int col = wn * 32 + nt * 16 + lm;
            #pragma unroll
            for (int j = 0; j < 8; ++j) {
                int kk = ((j < 4) ? (2 * j) : (16 + 2 * (j - 4))) + 8 * g;
                b_hi[nt].u[j] = *reinterpret_cast<const unsigned int*>(&sB_hi[col * BK + kk]);
                b_lo[nt].u[j] = *reinterpret_cast<const unsigned int*>(&sB_lo[col * BK + kk]);
            }
        }
        #pragma unroll
        for (int mt = 0; mt < 4; ++mt) {
            #pragma unroll
            for (int nt = 0; nt < 2; ++nt) {
                acc[mt][nt] = __builtin_amdgcn_wmma_f32_16x16x32_bf16(
                    false, a_hi[mt].v, false, b_hi[nt].v, (short)0, acc[mt][nt], false, false);
                acc[mt][nt] = __builtin_amdgcn_wmma_f32_16x16x32_bf16(
                    false, a_hi[mt].v, false, b_lo[nt].v, (short)0, acc[mt][nt], false, false);
                acc[mt][nt] = __builtin_amdgcn_wmma_f32_16x16x32_bf16(
                    false, a_lo[mt].v, false, b_hi[nt].v, (short)0, acc[mt][nt], false, false);
            }
        }
    }

    #pragma unroll
    for (int nt = 0; nt < 2; ++nt) {
        int n = n0 + wn * 32 + nt * 16 + lm;
        float bv = bc[n];
        #pragma unroll
        for (int mt = 0; mt < 4; ++mt) {
            #pragma unroll
            for (int j = 0; j < 8; ++j) {
                int m = m0 + wm * 64 + mt * 16 + 8 * g + j;
                Ob[(size_t)m * H_DIM + n] = acc[mt][nt][j] + bv;
            }
        }
    }
}

extern "C" void kernel_launch(void* const* d_in, const int* in_sizes, int n_in,
                              void* d_out, int out_size, void* d_ws, size_t ws_size,
                              hipStream_t stream) {
    const float* x    = (const float*)d_in[0];   // (32, 512, 1024) f32
    const int*   cats = (const int*)d_in[1];     // (32,) i32
    const float* W    = (const float*)d_in[2];   // (16, 1024, 4096) f32
    const float* bias = (const float*)d_in[3];   // (16, 4096) f32
    float*       out  = (float*)d_out;           // (32, 512, 4096) f32

    const size_t xElems = (size_t)NBATCH * S_DIM * I_DIM;     // 16 Mi
    const size_t wElems = (size_t)NCAT * I_DIM * H_DIM;       // 64 Mi
    const size_t need   = (2 * xElems + 2 * wElems) * sizeof(unsigned short); // 320 MB

    if (ws_size >= need) {
        unsigned short* Xhi  = (unsigned short*)d_ws;
        unsigned short* Xlo  = Xhi + xElems;
        unsigned short* WhiT = Xlo + xElems;
        unsigned short* WloT = WhiT + wElems;

        hipLaunchKernelGGL(convert_x_bf16x2, dim3((unsigned)(xElems / (256 * 4))), dim3(256),
                           0, stream, x, Xhi, Xlo);
        hipLaunchKernelGGL(convert_w_transpose_bf16x2, dim3(H_DIM / 32, I_DIM / 32, NCAT),
                           dim3(256), 0, stream, W, WhiT, WloT);
        hipLaunchKernelGGL(cat_linear_wmma_tdm, dim3(H_DIM / BN, S_DIM / BM, NBATCH),
                           dim3(256), 0, stream, Xhi, Xlo, WhiT, WloT, cats, bias, out);
    } else {
        hipLaunchKernelGGL(cat_linear_wmma_bf16x3, dim3(H_DIM / BN, S_DIM / BM, NBATCH),
                           dim3(256), 0, stream, x, cats, W, bias, out);
    }
}
